// ActorRecurrent_42253888258563
// MI455X (gfx1250) — matrix operand order repaired and checked by reading, not compiled, not run on hardware
//
#include <hip/hip_runtime.h>
#include <hip/hip_bf16.h>
#include <cmath>

// ---------------------------------------------------------------------------
// Problem constants (from reference)
// ---------------------------------------------------------------------------
#define BATCH   1024
#define TSTEPS  512
#define SDIM    64          // state dim (GRU input)
#define HID     256         // hidden
#define GATES   768         // 3*HID, gate order r,z,n
#define AH      256         // actor hidden
#define AD      8           // action dim

// Tiling
#define BM      32          // batch rows per workgroup (2 WMMA M-tiles)
#define GST     (GATES + 8) // f32 LDS stride for Gi/Gh (avoid 64-bank aliasing)
#define HFST    (HID + 4)   // f32 LDS stride for h master
#define HBST    (HID + 8)   // bf16 LDS stride for h WMMA copy

typedef __attribute__((ext_vector_type(16))) __bf16 v16bf;
typedef __attribute__((ext_vector_type(8)))  float  v8f;

// ---------------------------------------------------------------------------
// WMMA fragment helpers (CDNA5 ISA 7.12.2 lane layouts, wave32)
// ---------------------------------------------------------------------------
// A fragment: 16x32 bf16, src points at tile origin (row-major, stride ld).
// lanes 0-15: row M=lane, K = {0..7, 16..23}; lanes 16-31: same row, K+8.
__device__ __forceinline__ v16bf load_a_bf16(const __bf16* src, int ld, int lane) {
    const int r    = lane & 15;
    const int koff = (lane >> 4) << 3;          // 0 or 8
    const __bf16* p = src + r * ld + koff;
    v16bf a;
#pragma unroll
    for (int e = 0; e < 8; ++e) a[e] = p[e];
#pragma unroll
    for (int e = 0; e < 8; ++e) a[8 + e] = p[16 + e];
    return a;
}

// Same A layout but source is fp32 (x_seq), converted to bf16 on the fly.
__device__ __forceinline__ v16bf load_a_f32(const float* src, int ld, int lane) {
    const int r    = lane & 15;
    const int koff = (lane >> 4) << 3;
    const float* p = src + r * ld + koff;
    v16bf a;
#pragma unroll
    for (int e = 0; e < 8; ++e) a[e] = (__bf16)p[e];
#pragma unroll
    for (int e = 0; e < 8; ++e) a[8 + e] = (__bf16)p[16 + e];
    return a;
}

// B fragment: 32x16 bf16 with B[k][n] = W[n0+n][k0+k] (W row-major, stride ldw).
// w points at W + n0*ldw + k0. lanes 0-15: col N=lane, K = {0..7, 16..23};
// lanes 16-31: col N=lane-16, K+8 (matches A layout).
__device__ __forceinline__ v16bf load_b_w(const __bf16* w, int ldw, int lane) {
    const int n  = lane & 15;
    const int kb = (lane >> 4) << 3;            // 0 or 8
    const __bf16* p = w + n * ldw + kb;
    v16bf b;
#pragma unroll
    for (int e = 0; e < 8; ++e) b[e] = p[e];
#pragma unroll
    for (int e = 0; e < 8; ++e) b[8 + e] = p[16 + e];
    return b;
}

// C/D: 8 VGPRs; lanes 0-15: (M=v, N=lane); lanes 16-31: (M=v+8, N=lane-16).
__device__ __forceinline__ void store_c(float* dst, int ld, int lane, v8f c) {
    const int n  = lane & 15;
    const int mb = (lane >> 4) << 3;
#pragma unroll
    for (int v = 0; v < 8; ++v) dst[(mb + v) * ld + n] = c[v];
}

__device__ __forceinline__ float sigm(float x) { return 1.0f / (1.0f + __expf(-x)); }

// ---------------------------------------------------------------------------
// Kernel 0: fp32 -> bf16 weight conversion (one-time, tiny)
// ---------------------------------------------------------------------------
__global__ void cvt_bf16_kernel(const float* __restrict__ in, __bf16* __restrict__ out, int n) {
    int i = blockIdx.x * blockDim.x + threadIdx.x;
    if (i < n) out[i] = (__bf16)in[i];
}

// ---------------------------------------------------------------------------
// Kernel 1: persistent GRU recurrence. 32 WGs x 256 threads (8 waves).
// Each WG owns BM=32 batch rows; h lives in LDS for all 512 steps.
// ---------------------------------------------------------------------------
__global__ __launch_bounds__(256, 1)
void gru_recurrent_kernel(const float* __restrict__ x_seq,   // [B,T,64]
                          const __bf16* __restrict__ w_ih,   // [768,64] bf16
                          const __bf16* __restrict__ w_hh,   // [768,256] bf16
                          const float* __restrict__ b_ih,    // [768]
                          const float* __restrict__ b_hh,    // [768]
                          float* __restrict__ h_out)         // [B,256]
{
    extern __shared__ char smem[];
    float*  sGi = (float*)smem;                 // [BM][GST]  x@W_ih^T preacts
    float*  sGh = sGi + BM * GST;               // [BM][GST]  h@W_hh^T preacts
    float*  sHf = sGh + BM * GST;               // [BM][HFST] h master (fp32)
    __bf16* sHb = (__bf16*)(sHf + BM * HFST);   // [BM][HBST] h WMMA copy (bf16)

    const int tid  = threadIdx.x;
    const int lane = tid & 31;                  // wave32
    const int wave = tid >> 5;
    const int mt   = wave >> 2;                 // M-tile: 0 or 1
    const int nt0  = (wave & 3) * 12;           // 12 N-tiles per wave (48 total)
    const int row_g0 = blockIdx.x * BM;         // first global batch row

    // h0 = 0
    for (int i = tid; i < BM * HFST; i += 256) sHf[i] = 0.0f;
    for (int i = tid; i < BM * HBST; i += 256) sHb[i] = (__bf16)0.0f;
    __syncthreads();

    for (int t = 0; t < TSTEPS; ++t) {
        // ---- A fragments: x_t tile (K=64 -> 2 frags), h tile (K=256 -> 8) --
        const float* xrow =
            x_seq + (size_t)(row_g0 + mt * 16) * (TSTEPS * SDIM) + (size_t)t * SDIM;
        v16bf ax0 = load_a_f32(xrow,       TSTEPS * SDIM, lane);   // K 0..31
        v16bf ax1 = load_a_f32(xrow + 32,  TSTEPS * SDIM, lane);   // K 32..63

        v16bf ah[8];
#pragma unroll
        for (int kk = 0; kk < 8; ++kk)
            ah[kk] = load_a_bf16(sHb + (mt * 16) * HBST + kk * 32, HBST, lane);

        // ---- 12 N-tiles per wave: gi (2 wmma) + gh (8 wmma) each -----------
        for (int nti = 0; nti < 12; ++nti) {
            const int n0 = (nt0 + nti) * 16;

            v8f ci = {};
            {
                v16bf b0 = load_b_w(w_ih + (size_t)n0 * SDIM,      SDIM, lane);
                ci = __builtin_amdgcn_wmma_f32_16x16x32_bf16(false, ax0, false, b0,
                                                             (short)0, ci, false, false);
                v16bf b1 = load_b_w(w_ih + (size_t)n0 * SDIM + 32, SDIM, lane);
                ci = __builtin_amdgcn_wmma_f32_16x16x32_bf16(false, ax1, false, b1,
                                                             (short)0, ci, false, false);
            }
            v8f ch = {};
#pragma unroll
            for (int kk = 0; kk < 8; ++kk) {
                v16bf bw = load_b_w(w_hh + (size_t)n0 * HID + kk * 32, HID, lane);
                ch = __builtin_amdgcn_wmma_f32_16x16x32_bf16(false, ah[kk], false, bw,
                                                             (short)0, ch, false, false);
            }
            store_c(sGi + (mt * 16) * GST + n0, GST, lane, ci);
            store_c(sGh + (mt * 16) * GST + n0, GST, lane, ch);
        }
        __syncthreads();

        // ---- gate fusion: 8192 elements, 32 per thread ---------------------
        for (int i = tid; i < BM * HID; i += 256) {
            const int row = i >> 8;         // /256
            const int j   = i & 255;
            const float ir = sGi[row * GST +        j] + b_ih[       j];
            const float iz = sGi[row * GST + 256 +  j] + b_ih[256 +  j];
            const float in = sGi[row * GST + 512 +  j] + b_ih[512 +  j];
            const float hr = sGh[row * GST +        j] + b_hh[       j];
            const float hz = sGh[row * GST + 256 +  j] + b_hh[256 +  j];
            const float hn = sGh[row * GST + 512 +  j] + b_hh[512 +  j];
            const float r  = sigm(ir + hr);
            const float z  = sigm(iz + hz);
            const float n  = tanhf(in + r * hn);
            const float hp = sHf[row * HFST + j];
            const float hnew = (1.0f - z) * n + z * hp;
            sHf[row * HFST + j] = hnew;
            sHb[row * HBST + j] = (__bf16)hnew;
        }

        // warm L2/WGP$ for next timestep's x slice (global_prefetch_b8)
        if (t + 1 < TSTEPS) {
            const float* nx = x_seq + (size_t)(row_g0 + (tid >> 3)) * (TSTEPS * SDIM)
                                    + (size_t)(t + 1) * SDIM + (tid & 7) * 8;
            __builtin_prefetch(nx, 0, 1);
        }
        __syncthreads();
    }

    // final hidden state to workspace
    for (int i = tid; i < BM * HID; i += 256) {
        const int row = i >> 8, j = i & 255;
        h_out[(size_t)(row_g0 + row) * HID + j] = sHf[row * HFST + j];
    }
}

// ---------------------------------------------------------------------------
// Kernel 2: actor head. 64 WGs x 256 threads; WMMA for 256x256 base layer,
// plain fp32 dots for the 8-wide dir/mag heads, fused tanh*sigmoid.
// ---------------------------------------------------------------------------
__global__ __launch_bounds__(256, 1)
void actor_head_kernel(const float* __restrict__ h_in,        // [B,256]
                       const __bf16* __restrict__ w_base_bf,  // [256,256] bf16
                       const float* __restrict__ b_base,      // [256]
                       const float* __restrict__ w_dir,       // [8,256]
                       const float* __restrict__ b_dir,       // [8]
                       const float* __restrict__ w_mag,       // [8,256]
                       const float* __restrict__ b_mag,       // [8]
                       float* __restrict__ out)               // [B,8]
{
    __shared__ __bf16 sH[16 * HBST];
    __shared__ float  sB[16 * HFST];
    __shared__ float  sD[256];

    const int tid  = threadIdx.x;
    const int lane = tid & 31;
    const int wave = tid >> 5;
    const int row0 = blockIdx.x * 16;

    // stage 16x256 h tile as bf16
    for (int i = tid; i < 16 * HID; i += 256) {
        const int r = i >> 8, j = i & 255;
        sH[r * HBST + j] = (__bf16)h_in[(size_t)(row0 + r) * HID + j];
    }
    __syncthreads();

    // base layer: 16 N-tiles, 2 per wave, K=256 (8 wmma each)
    v16bf ah[8];
#pragma unroll
    for (int kk = 0; kk < 8; ++kk) ah[kk] = load_a_bf16(sH + kk * 32, HBST, lane);

    for (int nti = 0; nti < 2; ++nti) {
        const int n0 = (wave * 2 + nti) * 16;
        v8f c = {};
#pragma unroll
        for (int kk = 0; kk < 8; ++kk) {
            v16bf bw = load_b_w(w_base_bf + (size_t)n0 * AH + kk * 32, AH, lane);
            c = __builtin_amdgcn_wmma_f32_16x16x32_bf16(false, ah[kk], false, bw,
                                                        (short)0, c, false, false);
        }
        const int n = lane & 15, mb = (lane >> 4) << 3;
#pragma unroll
        for (int v = 0; v < 8; ++v) {
            const float val = c[v] + b_base[n0 + n];
            sB[(mb + v) * HFST + n0 + n] = val > 0.0f ? val : 0.0f;   // ReLU
        }
    }
    __syncthreads();

    // heads: tid = row*16 + act*2 + head  (16 rows x 8 actions x 2 heads)
    const int head = tid & 1, act = (tid >> 1) & 7, row = tid >> 4;
    const float* w = head ? w_mag : w_dir;
    float acc = head ? b_mag[act] : b_dir[act];
#pragma unroll 8
    for (int k = 0; k < HID; ++k) acc += sB[row * HFST + k] * w[act * HID + k];
    sD[tid] = acc;
    __syncthreads();

    if (tid < 128) {
        const int row2 = tid >> 3, act2 = tid & 7;
        const float d = sD[(row2 << 4) | (act2 << 1)];
        const float m = sD[(row2 << 4) | (act2 << 1) | 1];
        out[(size_t)(row0 + row2) * AD + act2] = tanhf(d) * sigm(m);
    }
}

// ---------------------------------------------------------------------------
// Host side
// ---------------------------------------------------------------------------
extern "C" void kernel_launch(void* const* d_in, const int* in_sizes, int n_in,
                              void* d_out, int out_size, void* d_ws, size_t ws_size,
                              hipStream_t stream) {
    (void)in_sizes; (void)n_in; (void)out_size; (void)ws_size;

    const float* x_seq  = (const float*)d_in[0];
    const float* w_ih   = (const float*)d_in[1];   // [768,64]
    const float* w_hh   = (const float*)d_in[2];   // [768,256]
    const float* b_ih   = (const float*)d_in[3];
    const float* b_hh   = (const float*)d_in[4];
    const float* w_base = (const float*)d_in[5];   // [256,256]
    const float* b_base = (const float*)d_in[6];
    const float* w_dir  = (const float*)d_in[7];   // [8,256]
    const float* b_dir  = (const float*)d_in[8];
    const float* w_mag  = (const float*)d_in[9];   // [8,256]
    const float* b_mag  = (const float*)d_in[10];
    float* out = (float*)d_out;

    // workspace layout (all 256B-aligned)
    char* ws = (char*)d_ws;
    __bf16* w_ih_bf   = (__bf16*)ws;                         ws += ((size_t)GATES * SDIM * 2 + 255) & ~255ull;
    __bf16* w_hh_bf   = (__bf16*)ws;                         ws += ((size_t)GATES * HID  * 2 + 255) & ~255ull;
    __bf16* w_base_bf = (__bf16*)ws;                         ws += ((size_t)AH    * AH   * 2 + 255) & ~255ull;
    float*  h_final   = (float*)ws;                          // [B,256] fp32

    // 0) one-time weight conversion to bf16
    {
        const int n1 = GATES * SDIM, n2 = GATES * HID, n3 = AH * AH;
        cvt_bf16_kernel<<<(n1 + 255) / 256, 256, 0, stream>>>(w_ih,   w_ih_bf,   n1);
        cvt_bf16_kernel<<<(n2 + 255) / 256, 256, 0, stream>>>(w_hh,   w_hh_bf,   n2);
        cvt_bf16_kernel<<<(n3 + 255) / 256, 256, 0, stream>>>(w_base, w_base_bf, n3);
    }

    // 1) persistent GRU recurrence: 32 WGs, 243 KB LDS each (needs CDNA5 320KB/WGP)
    {
        const size_t smem = (size_t)(2 * BM * GST) * sizeof(float)   // Gi + Gh
                          + (size_t)(BM * HFST)    * sizeof(float)   // h fp32
                          + (size_t)(BM * HBST)    * sizeof(__bf16); // h bf16
        gru_recurrent_kernel<<<BATCH / BM, 256, smem, stream>>>(
            x_seq, w_ih_bf, w_hh_bf, b_ih, b_hh, h_final);
    }

    // 2) actor head
    actor_head_kernel<<<BATCH / 16, 256, 0, stream>>>(
        h_final, w_base_bf, b_base, w_dir, b_dir, w_mag, b_mag, out);
}
